// Model_53395033424044
// MI455X (gfx1250) — matrix operand order repaired and checked by reading, not compiled, not run on hardware
//
#include <hip/hip_runtime.h>

// ---------------- problem sizes ----------------
#define B_  4096
#define T_  48
#define F_  64
#define H_  128
#define BT  128     // batch rows per block
#define NTH 256     // 8 waves of 32

typedef __attribute__((ext_vector_type(16))) __bf16 v16bf;
typedef __attribute__((ext_vector_type(8)))  __bf16 v8bf;
typedef __attribute__((ext_vector_type(8)))  float  v8f;

// ---------------- helpers ----------------
__device__ __forceinline__ unsigned short f2bf(float f) {
    unsigned int u = __builtin_bit_cast(unsigned int, f);
    u += 0x7FFFu + ((u >> 16) & 1u);           // round-to-nearest-even
    return (unsigned short)(u >> 16);
}
__device__ __forceinline__ float bf2f(unsigned short u) {
    unsigned int x = ((unsigned int)u) << 16;
    return __builtin_bit_cast(float, x);
}
__device__ __forceinline__ float sigmoidf_(float x) {
    return 1.0f / (1.0f + __expf(-x));
}

// A operand (16x32 bf16) from LDS row-major [rows][ldk]
__device__ __forceinline__ v16bf load_a(const unsigned short* base, int ldk,
                                        int rowBase, int kt) {
    int lane = threadIdx.x & 31;
    const unsigned short* p = base + (rowBase + (lane & 15)) * ldk
                                   + kt + ((lane >> 4) << 3);
    v8bf lo = *(const v8bf*)p;          // K = kt+koff .. +7
    v8bf hi = *(const v8bf*)(p + 16);   // K = kt+koff+16 .. +23
    return __builtin_shufflevector(lo, hi, 0,1,2,3,4,5,6,7,8,9,10,11,12,13,14,15);
}
// B operand (32x16 bf16) from global weights row-major [N][K] (i.e. W[out][in])
__device__ __forceinline__ v16bf load_b(const unsigned short* W, int K,
                                        int nBase, int kt) {
    int lane = threadIdx.x & 31;
    const unsigned short* p = W + (size_t)(nBase + (lane & 15)) * K
                                + kt + ((lane >> 4) << 3);
    v8bf lo = *(const v8bf*)p;          // K = kt+koff .. +7
    v8bf hi = *(const v8bf*)(p + 16);   // K = kt+koff+16 .. +23
    return __builtin_shufflevector(lo, hi, 0,1,2,3,4,5,6,7,8,9,10,11,12,13,14,15);
}
#define WMMA_BF16(acc, a, b) \
    (acc) = __builtin_amdgcn_wmma_f32_16x16x32_bf16(false, (a), false, (b), \
                                                    (short)0, (acc), false, false)

// ---------------- ws layout ----------------
// floats: [0..48)   denoms
//         [64..576) bias_g = b_ih + b_hh
// ushorts @ byte 2304: Wdh(8192) Wh(8192) Wf(4096,diag=0) Wc(8192) Wih(65536) Whh(65536)
#define WBF_BYTE_OFF 2304
#define O_WDH 0
#define O_WH  8192
#define O_WF  16384
#define O_WC  20480
#define O_WIH 28672
#define O_WHH 94208
#define N_WBF 159744

// ---------------- pre-pass kernels ----------------
__global__ void init_kernel(float* out_loss) {
    if (threadIdx.x == 0 && blockIdx.x == 0) *out_loss = 0.0f;
}

__global__ void denom_kernel(const float* __restrict__ masks, float* denoms) {
    int t = blockIdx.x;
    float s = 0.f;
    for (int idx = threadIdx.x; idx < B_ * F_; idx += blockDim.x) {
        int b = idx >> 6, f = idx & 63;
        s += masks[(size_t)b * T_ * F_ + (size_t)t * F_ + f];
    }
    __shared__ float red[256];
    red[threadIdx.x] = s; __syncthreads();
    for (int o = 128; o > 0; o >>= 1) {
        if (threadIdx.x < o) red[threadIdx.x] += red[threadIdx.x + o];
        __syncthreads();
    }
    if (threadIdx.x == 0) denoms[t] = red[0] + 1e-5f;
}

__global__ void prep_kernel(const float* Wdh, const float* Wh, const float* Wf,
                            const float* Wc, const float* Wih, const float* Whh,
                            const float* b_ih, const float* b_hh,
                            float* ws_f, unsigned short* wbf) {
    int i = blockIdx.x * blockDim.x + threadIdx.x;
    for (int idx = i; idx < N_WBF; idx += gridDim.x * blockDim.x) {
        float v;
        if      (idx < O_WH)  v = Wdh[idx - O_WDH];
        else if (idx < O_WF)  v = Wh[idx - O_WH];
        else if (idx < O_WC)  { int j = idx - O_WF; int r = j >> 6, c = j & 63;
                                v = (r == c) ? 0.0f : Wf[j]; }
        else if (idx < O_WIH) v = Wc[idx - O_WC];
        else if (idx < O_WHH) v = Wih[idx - O_WIH];
        else                  v = Whh[idx - O_WHH];
        wbf[idx] = f2bf(v);
    }
    if (i < 512) ws_f[64 + i] = b_ih[i] + b_hh[i];
}

// ---------------- main persistent scan kernel ----------------
__global__ __launch_bounds__(NTH, 1)
void brits_scan_kernel(const float* __restrict__ values,
                       const float* __restrict__ masks,
                       const float* __restrict__ deltas,
                       const float* __restrict__ bdh, const float* __restrict__ Wdx,
                       const float* __restrict__ bdx, const float* __restrict__ bh,
                       const float* __restrict__ bfb, const float* __restrict__ bc,
                       const float* __restrict__ Wo,  const float* __restrict__ bo,
                       const float* __restrict__ ws_f,
                       const unsigned short* __restrict__ wbf,
                       float* out_loss, float* __restrict__ out_imp,
                       float* __restrict__ out_pred) {
    // -------- LDS staging (~293 KB of the 320 KB WGP budget) --------
    __shared__ unsigned short d_bf  [BT * F_];       // delta bf16 (A for gamma_h)
    __shared__ unsigned short h_bf  [BT * H_];       // decayed h bf16 (A for x_h, gates)
    __shared__ unsigned short hn_bf [BT * H_];       // post-LSTM h state (bf16)
    __shared__ unsigned short xc_bf [BT * F_];       // x_c bf16 (A for z_h)
    __shared__ unsigned short gxm_bf[BT * 2 * F_];   // [gamma_x | m] (A for alpha)
    __shared__ unsigned short ccm_bf[BT * 2 * F_];   // [c_c | m]     (A for gates)
    __shared__ float xh_s[BT * F_];                  // x_h f32
    __shared__ float x_s [BT * F_];                  // x f32
    __shared__ float c_s [BT * H_];                  // LSTM cell state f32
    __shared__ float p_wdx[F_], p_bdx[F_], p_bh[F_], p_bf[F_], p_bc[F_];
    __shared__ float p_bdh[H_], p_bg[4 * H_];
    __shared__ float lred[NTH];

    const int tid  = threadIdx.x;
    const int lane = tid & 31;
    const int lo = lane & 15, hi = lane >> 4;
    const int rowBase = (tid >> 5) * 16;             // wave's 16-row slab
    const size_t bG0 = (size_t)blockIdx.x * BT;

    for (int i = tid; i < F_; i += NTH) {
        p_wdx[i] = Wdx[i * F_ + i]; p_bdx[i] = bdx[i];
        p_bh[i] = bh[i]; p_bf[i] = bfb[i]; p_bc[i] = bc[i];
    }
    for (int i = tid; i < H_;     i += NTH) p_bdh[i] = bdh[i];
    for (int i = tid; i < 4 * H_; i += NTH) p_bg[i]  = ws_f[64 + i];
    for (int i = tid; i < BT * H_; i += NTH) { c_s[i] = 0.f; hn_bf[i] = 0; }

    const unsigned short* Wdh_bf = wbf + O_WDH;
    const unsigned short* Wh_bf  = wbf + O_WH;
    const unsigned short* Wf_bf  = wbf + O_WF;
    const unsigned short* Wc_bf  = wbf + O_WC;
    const unsigned short* Wih_bf = wbf + O_WIH;
    const unsigned short* Whh_bf = wbf + O_WHH;

    float loss_acc = 0.f;
    __syncthreads();

    for (int t = 0; t < T_; ++t) {
        const float invd = 1.0f / ws_f[t];
        float lstep = 0.f;

        // ---- P0: elementwise loads, gamma_x, bf16 staging ----
        {
            int row = tid >> 1;
            int cb  = (tid & 1) << 5;
            size_t g = (bG0 + row) * (size_t)(T_ * F_) + (size_t)t * F_;
            #pragma unroll
            for (int j = 0; j < 32; ++j) {
                int f = cb + j;
                float d = deltas[g + f];
                float m = masks [g + f];
                float x = values[g + f];
                float gx = __expf(-fmaxf(d * p_wdx[f] + p_bdx[f], 0.f));
                d_bf  [row * F_ + f]            = f2bf(d);
                unsigned short mb               = f2bf(m);   // masks are 0/1: exact
                gxm_bf[row * 2 * F_ + f]        = f2bf(gx);
                gxm_bf[row * 2 * F_ + F_ + f]   = mb;
                ccm_bf[row * 2 * F_ + F_ + f]   = mb;
                x_s   [row * F_ + f]            = x;
            }
            if (t + 1 < T_) {                       // global_prefetch_b8 next step
                __builtin_prefetch(deltas + g + F_ + cb, 0, 0);
                __builtin_prefetch(masks  + g + F_ + cb, 0, 0);
                __builtin_prefetch(values + g + F_ + cb, 0, 0);
            }
        }
        __syncthreads();

        // ---- P1: gamma_h = exp(-relu(d Wdh^T + bdh)); h = h_prev * gamma_h ----
        #pragma unroll 2
        for (int nt = 0; nt < 8; ++nt) {
            v8f acc = {};
            #pragma unroll
            for (int kt = 0; kt < 2; ++kt)
                WMMA_BF16(acc, load_a(d_bf, F_, rowBase, kt * 32),
                               load_b(Wdh_bf, F_, nt * 16, kt * 32));
            int col = nt * 16 + lo;
            float bias = p_bdh[col];
            #pragma unroll
            for (int v = 0; v < 8; ++v) {
                float gmh = __expf(-fmaxf(acc[v] + bias, 0.f));
                int idx = (rowBase + v + 8 * hi) * H_ + col;
                float hd = bf2f(hn_bf[idx]) * gmh;
                h_bf[idx] = f2bf(hd);
            }
        }
        __syncthreads();

        // ---- P2: x_h = h Wh^T + bh; loss1; x_c ----
        #pragma unroll 2
        for (int nt = 0; nt < 4; ++nt) {
            v8f acc = {};
            #pragma unroll
            for (int kt = 0; kt < 4; ++kt)
                WMMA_BF16(acc, load_a(h_bf, H_, rowBase, kt * 32),
                               load_b(Wh_bf, H_, nt * 16, kt * 32));
            int col = nt * 16 + lo;
            float bias = p_bh[col];
            #pragma unroll
            for (int v = 0; v < 8; ++v) {
                int row = rowBase + v + 8 * hi;
                float xh = acc[v] + bias;
                xh_s[row * F_ + col] = xh;
                float x = x_s[row * F_ + col];
                float m = bf2f(gxm_bf[row * 2 * F_ + F_ + col]);
                lstep += fabsf(x - xh) * m;
                xc_bf[row * F_ + col] = f2bf(m * x + (1.f - m) * xh);
            }
        }
        __syncthreads();

        // ---- P3: z_h & alpha (fused), c_h, loss2/3, c_c -> imputations ----
        #pragma unroll 1
        for (int nt = 0; nt < 4; ++nt) {
            v8f accz = {};
            #pragma unroll
            for (int kt = 0; kt < 2; ++kt)
                WMMA_BF16(accz, load_a(xc_bf, F_, rowBase, kt * 32),
                                load_b(Wf_bf, F_, nt * 16, kt * 32));
            v8f acca = {};
            #pragma unroll
            for (int kt = 0; kt < 4; ++kt)
                WMMA_BF16(acca, load_a(gxm_bf, 2 * F_, rowBase, kt * 32),
                                load_b(Wc_bf, 2 * F_, nt * 16, kt * 32));
            int col = nt * 16 + lo;
            float bz = p_bf[col], ba = p_bc[col];
            #pragma unroll
            for (int v = 0; v < 8; ++v) {
                int row = rowBase + v + 8 * hi;
                float zh = accz[v] + bz;
                float al = acca[v] + ba;
                float x  = x_s [row * F_ + col];
                float m  = bf2f(gxm_bf[row * 2 * F_ + F_ + col]);
                float xh = xh_s[row * F_ + col];
                lstep += fabsf(x - zh) * m;
                float ch = al * zh + (1.f - al) * xh;
                lstep += fabsf(x - ch) * m;
                float cc = m * x + (1.f - m) * ch;
                out_imp[(bG0 + row) * (size_t)(T_ * F_) + (size_t)t * F_ + col] = cc;
                ccm_bf[row * 2 * F_ + col] = f2bf(cc);
            }
        }
        __syncthreads();

        // ---- P4: gates = [c_c|m] Wih^T + h Whh^T + bias; LSTM update ----
        #pragma unroll 1
        for (int ht = 0; ht < 8; ++ht) {
            float gv[4][8];
            #pragma unroll
            for (int g = 0; g < 4; ++g) {            // i, f, g, o
                int nb = g * H_ + ht * 16;
                v8f acc = {};
                #pragma unroll
                for (int kt = 0; kt < 4; ++kt)
                    WMMA_BF16(acc, load_a(ccm_bf, 2 * F_, rowBase, kt * 32),
                                   load_b(Wih_bf, 2 * F_, nb, kt * 32));
                #pragma unroll
                for (int kt = 0; kt < 4; ++kt)
                    WMMA_BF16(acc, load_a(h_bf, H_, rowBase, kt * 32),
                                   load_b(Whh_bf, H_, nb, kt * 32));
                float bias = p_bg[nb + lo];
                #pragma unroll
                for (int v = 0; v < 8; ++v) gv[g][v] = acc[v] + bias;
            }
            int col = ht * 16 + lo;
            #pragma unroll
            for (int v = 0; v < 8; ++v) {
                int idx = (rowBase + v + 8 * hi) * H_ + col;
                float ig = sigmoidf_(gv[0][v]);
                float fg = sigmoidf_(gv[1][v]);
                float gg = tanhf(gv[2][v]);
                float og = sigmoidf_(gv[3][v]);
                float cn = fg * c_s[idx] + ig * gg;
                c_s[idx] = cn;
                hn_bf[idx] = f2bf(og * tanhf(cn));
            }
        }
        loss_acc += lstep * invd;
        __syncthreads();   // protect LDS buffers before next step's P0
    }

    // ---- predictions = h Wo^T + bo (final h lives in hn_bf) ----
    if (tid < BT) {
        float s = 0.f;
        #pragma unroll 4
        for (int k = 0; k < H_; ++k)
            s += bf2f(hn_bf[tid * H_ + k]) * Wo[k];
        out_pred[bG0 + tid] = s + bo[0];
    }

    // ---- loss reduction ----
    lred[tid] = loss_acc; __syncthreads();
    for (int o = NTH / 2; o > 0; o >>= 1) {
        if (tid < o) lred[tid] += lred[tid + o];
        __syncthreads();
    }
    if (tid == 0) atomicAdd(out_loss, lred[0]);
}

// ---------------- launch ----------------
extern "C" void kernel_launch(void* const* d_in, const int* in_sizes, int n_in,
                              void* d_out, int out_size, void* d_ws, size_t ws_size,
                              hipStream_t stream) {
    const float* values = (const float*)d_in[0];
    const float* masks  = (const float*)d_in[1];
    const float* deltas = (const float*)d_in[2];
    const float* Wdh  = (const float*)d_in[3];
    const float* bdh  = (const float*)d_in[4];
    const float* Wdx  = (const float*)d_in[5];
    const float* bdx  = (const float*)d_in[6];
    const float* Wh   = (const float*)d_in[7];
    const float* bh   = (const float*)d_in[8];
    const float* Wf   = (const float*)d_in[9];
    const float* bfb  = (const float*)d_in[10];
    const float* Wc   = (const float*)d_in[11];
    const float* bc   = (const float*)d_in[12];
    const float* Wih  = (const float*)d_in[13];
    const float* Whh  = (const float*)d_in[14];
    const float* b_ih = (const float*)d_in[15];
    const float* b_hh = (const float*)d_in[16];
    const float* Wo   = (const float*)d_in[17];
    const float* bo   = (const float*)d_in[18];

    float* ws_f = (float*)d_ws;
    unsigned short* wbf = (unsigned short*)((char*)d_ws + WBF_BYTE_OFF);

    float* out       = (float*)d_out;
    float* out_loss  = out;                                   // [1]
    float* out_imp   = out + 1;                               // [B,T,F]
    float* out_pred  = out + 1 + (size_t)B_ * T_ * F_;        // [B,1]

    init_kernel <<<1, 1, 0, stream>>>(out_loss);
    denom_kernel<<<T_, 256, 0, stream>>>(masks, ws_f);
    prep_kernel <<<256, 256, 0, stream>>>(Wdh, Wh, Wf, Wc, Wih, Whh,
                                          b_ih, b_hh, ws_f, wbf);
    brits_scan_kernel<<<B_ / BT, NTH, 0, stream>>>(
        values, masks, deltas, bdh, Wdx, bdx, bh, bfb, bc, Wo, bo,
        ws_f, wbf, out_loss, out_imp, out_pred);
}
